// KGFlowBlock_79250736545902
// MI455X (gfx1250) — compile-verified, matrix operand order checked by reading, not run on hardware
//
#include <hip/hip_runtime.h>
#include <hip/hip_bf16.h>
#include <math.h>

typedef __attribute__((ext_vector_type(16))) _Float16 v16h;
typedef __attribute__((ext_vector_type(8)))  _Float16 v8h;
typedef __attribute__((ext_vector_type(8)))  float    v8f;

#define DEV __device__ __forceinline__

static constexpr int BSZ   = 2;
static constexpr int NREG  = 1024;
static constexpr int NHOUR = 24;
static constexpr int DIM   = 64;
static constexpr int NRel  = 6;
static constexpr int NE    = 32768;
static constexpr int GG    = BSZ * NHOUR;          // 48
static constexpr int NT    = BSZ * NREG * NHOUR;   // 49152 tokens
static constexpr int N2    = BSZ * NREG;           // 2048
static constexpr int TDIM  = 128;
static constexpr int NHEAD = 4;
static constexpr int HD    = 16;
static constexpr int MT    = 4;                    // m-tiles per block in gemm

// ---------------------------------------------------------------------------
// WMMA helpers. CDNA5 16-bit A fragment (16x32, wave32):
//   lanes 0-15  : row M=lane,    elems 0-7 => K=kk+0..7,  elems 8-15 => K=kk+16..23
//   lanes 16-31 : row M=lane-16, elems 0-7 => K=kk+8..15, elems 8-15 => K=kk+24..31
// B loaded column-per-lane from Bt (N x K row-major) with the same K mapping.
// C/D f32: elem v => row m0 + v + 8*(lane>=16), col n0 + lane%16.
// ---------------------------------------------------------------------------
DEV v16h load_frag(const _Float16* __restrict__ row, int kk, int hi) {
  v8h lo = *(const v8h*)(row + kk + hi * 8);
  v8h hh = *(const v8h*)(row + kk + 16 + hi * 8);
  return __builtin_shufflevector(lo, hh, 0, 1, 2, 3, 4, 5, 6, 7,
                                         8, 9, 10, 11, 12, 13, 14, 15);
}

DEV v8f wmma_f16(v16h a, v16h b, v8f c) {
  return __builtin_amdgcn_wmma_f32_16x16x32_f16(false, a, false, b,
                                                (short)0, c, false, false);
}

DEV void async_copy_b128_to_lds(const _Float16* gsrc, _Float16* lds_dst) {
  uint64_t ga = (uint64_t)(uintptr_t)gsrc;
  uint32_t la = (uint32_t)(uintptr_t)lds_dst;   // low 32 bits = LDS offset
  asm volatile("global_load_async_to_lds_b128 %0, %1, off"
               :: "v"(la), "v"(ga) : "memory");
}

// ---------------------------------------------------------------------------
// Generic f16 GEMM: C(MxN) = A(MxK) * Bt(NxK)^T  (+bias)(+resid)(+relu)
// Block = 64 threads (2 waves). Each block owns a 64-column panel of Bt,
// async-staged into LDS once, then amortized over MT=4 m-tiles. Each wave
// computes a 16x32 tile (2 accumulators share one A fragment).
// Requires M%64==0, N%64==0, K%32==0. Dynamic LDS = 64*K*2 bytes.
// ---------------------------------------------------------------------------
__global__ void gemm_f16_kernel(const _Float16* __restrict__ A,
                                const _Float16* __restrict__ Bt,
                                const float* __restrict__ bias,
                                const float* __restrict__ resid,
                                float* __restrict__ Cf,
                                _Float16* __restrict__ Ch,
                                int M, int N, int K, int relu) {
  extern __shared__ _Float16 smem[];             // 64 rows x K halfs
  int tid  = threadIdx.x;                        // 0..63
  int wave = tid >> 5;                           // 0..1
  int lane = tid & 31;
  int nPanel = blockIdx.y << 6;                  // 64 columns per block
  int mBase  = blockIdx.x * (16 * MT);

  // ---- async stage of the Bt panel into LDS (ASYNCcnt path) ----
  {
    const _Float16* src = Bt + (size_t)nPanel * K;
    int chunks = (64 * K) >> 3;                  // 16-byte chunks
    for (int i = tid; i < chunks; i += 64)
      async_copy_b128_to_lds(src + (size_t)i * 8, smem + (size_t)i * 8);
    asm volatile("s_wait_asynccnt 0" ::: "memory");
    __syncthreads();
  }

  int lid = lane & 15;
  int hi  = lane >> 4;
  const _Float16* b0row = smem + (size_t)((wave << 5) + lid) * K;
  const _Float16* b1row = b0row + (size_t)16 * K;
  int col0 = nPanel + (wave << 5) + lid;
  int col1 = col0 + 16;
  float bv0 = bias ? bias[col0] : 0.0f;
  float bv1 = bias ? bias[col1] : 0.0f;

  for (int mt = 0; mt < MT; ++mt) {
    int m0 = mBase + mt * 16;
    const _Float16* arow = A + (size_t)(m0 + lid) * K;
    if (mt + 1 < MT) __builtin_prefetch(arow + (size_t)16 * K, 0, 3);
    v8f c0 = {0.f, 0.f, 0.f, 0.f, 0.f, 0.f, 0.f, 0.f};
    v8f c1 = {0.f, 0.f, 0.f, 0.f, 0.f, 0.f, 0.f, 0.f};
    for (int kk = 0; kk < K; kk += 32) {
      v16h a  = load_frag(arow, kk, hi);
      v16h b0 = load_frag(b0row, kk, hi);
      v16h b1 = load_frag(b1row, kk, hi);
      c0 = wmma_f16(a, b0, c0);
      c1 = wmma_f16(a, b1, c1);
    }
#pragma unroll
    for (int v = 0; v < 8; ++v) {
      int row = m0 + (hi << 3) + v;
      size_t o0 = (size_t)row * N + col0;
      size_t o1 = (size_t)row * N + col1;
      float v0 = c0[v] + bv0;
      float v1 = c1[v] + bv1;
      if (resid) { v0 += resid[o0]; v1 += resid[o1]; }
      if (relu)  { v0 = fmaxf(v0, 0.0f); v1 = fmaxf(v1, 0.0f); }
      if (Cf) { Cf[o0] = v0; Cf[o1] = v1; }
      if (Ch) { Ch[o0] = (_Float16)v0; Ch[o1] = (_Float16)v1; }
    }
  }
}

// ---------------------------------------------------------------------------
// RGCN fused GEMM: h[m,:] = relu( sum_r aggN[m,r,:]@relW[r] + xg[m,:]@rootW + b )
// m flattens (g, n). Output written f16 at the out_spatial permutation.
// ---------------------------------------------------------------------------
__global__ void rgcn_gemm_kernel(const float* __restrict__ aggN,      // (G*NREG, NRel, 64) f32
                                 const _Float16* __restrict__ xgH,    // (G*NREG, 64)
                                 const _Float16* __restrict__ relT,   // (NRel, 64, 64) = W^T
                                 const _Float16* __restrict__ rootT,  // (64, 64) = W^T
                                 const float* __restrict__ bias,
                                 _Float16* __restrict__ osH) {        // (NT, 64)
  int wave = threadIdx.x >> 5;
  int lane = threadIdx.x & 31;
  int m0 = blockIdx.x << 4;
  int n0 = wave << 4;
  int lid = lane & 15;
  int hi  = lane >> 4;
  int mrow = m0 + lid;
  v8f c = {0.f, 0.f, 0.f, 0.f, 0.f, 0.f, 0.f, 0.f};
  for (int r = 0; r < NRel; ++r) {
    const float* af = aggN + ((size_t)mrow * NRel + r) * DIM;
    const _Float16* brow = relT + ((size_t)r * DIM + n0 + lid) * DIM;
    for (int kk = 0; kk < DIM; kk += 32) {
      v16h a;
#pragma unroll
      for (int j = 0; j < 8; ++j) {
        a[j]     = (_Float16)af[kk + hi * 8 + j];
        a[j + 8] = (_Float16)af[kk + 16 + hi * 8 + j];
      }
      v16h b = load_frag(brow, kk, hi);
      c = wmma_f16(a, b, c);
    }
  }
  {
    const _Float16* arow = xgH + (size_t)mrow * DIM;
    const _Float16* brow = rootT + (size_t)(n0 + lid) * DIM;
    for (int kk = 0; kk < DIM; kk += 32) {
      v16h a = load_frag(arow, kk, hi);
      v16h b = load_frag(brow, kk, hi);
      c = wmma_f16(a, b, c);
    }
  }
  int col = n0 + lid;
  float bv = bias[col];
#pragma unroll
  for (int v = 0; v < 8; ++v) {
    int m = m0 + (hi << 3) + v;
    int g = m >> 10;          // NREG == 1024
    int n = m & 1023;
    int b = g / NHOUR, hr = g % NHOUR;
    size_t tok = ((size_t)(b * NREG + n)) * NHOUR + hr;
    float val = fmaxf(c[v] + bv, 0.0f);
    osH[tok * DIM + col] = (_Float16)val;
  }
}

// ---------------------------------------------------------------------------
// Small scalar kernels
// ---------------------------------------------------------------------------
__global__ void zero_kernel(float* __restrict__ p, size_t n) {
  size_t stride = (size_t)gridDim.x * blockDim.x;
  for (size_t i = blockIdx.x * (size_t)blockDim.x + threadIdx.x; i < n; i += stride)
    p[i] = 0.0f;
}

__global__ void tmlp_kernel(const float* __restrict__ te, const float* __restrict__ w,
                            const float* __restrict__ b, float* __restrict__ t) {
  int idx = blockIdx.x * blockDim.x + threadIdx.x;
  if (idx >= BSZ * NHOUR * DIM) return;            // 2 * 1536
  int o = idx % (NHOUR * DIM);
  int bb = idx / (NHOUR * DIM);
  float acc = b[o];
  for (int k = 0; k < TDIM; ++k) {
    float x = te[bb * TDIM + k];
    float s = x / (1.0f + expf(-x));               // silu
    acc += s * w[k * (NHOUR * DIM) + o];
  }
  t[idx] = acc;
}

__global__ void add_time_kernel(const float* __restrict__ x_in, const float* __restrict__ t,
                                float* __restrict__ ot, _Float16* __restrict__ otH,
                                _Float16* __restrict__ xgH) {
  int idx = blockIdx.x * blockDim.x + threadIdx.x;
  if (idx >= NT * DIM) return;
  int d   = idx & 63;
  int tok = idx >> 6;
  int hr  = tok % NHOUR;
  int n   = (tok / NHOUR) % NREG;
  int b   = tok / (NHOUR * NREG);
  float v = x_in[idx] + t[(b * NHOUR + hr) * DIM + d];
  ot[idx]  = v;
  otH[idx] = (_Float16)v;
  xgH[(((size_t)(b * NHOUR + hr)) * NREG + n) * DIM + d] = (_Float16)v;
}

__global__ void cnt_kernel(const int* __restrict__ ei, const int* __restrict__ et,
                           float* __restrict__ cnt) {
  int e = blockIdx.x * blockDim.x + threadIdx.x;
  if (e >= NE) return;
  int dst = ei[NE + e];
  atomicAdd(&cnt[dst * NRel + et[e]], 1.0f);
}

__global__ void scatter_kernel(const int* __restrict__ ei, const int* __restrict__ et,
                               const float* __restrict__ ot, float* __restrict__ agg) {
  int e = blockIdx.x;
  int src = ei[e];
  int dst = ei[NE + e];
  int r   = et[e];
  for (int i = threadIdx.x; i < GG * DIM; i += blockDim.x) {
    int g = i >> 6;
    int d = i & 63;
    int b = g / NHOUR, hr = g % NHOUR;
    float v = ot[(((size_t)(b * NREG + src)) * NHOUR + hr) * DIM + d];
    atomicAdd(&agg[(((size_t)g * NREG + dst) * NRel + r) * DIM + d], v);
  }
}

__global__ void norm_kernel(float* __restrict__ agg, const float* __restrict__ cnt) {
  size_t idx = blockIdx.x * (size_t)blockDim.x + threadIdx.x;
  if (idx >= (size_t)GG * NREG * NRel * DIM) return;
  int seg = (int)((idx / DIM) % (NREG * NRel));
  agg[idx] /= fmaxf(cnt[seg], 1.0f);
}

__global__ void attn_temporal_kernel(const float* __restrict__ qkv,   // (NT, 192)
                                     _Float16* __restrict__ outH) {   // (NT, 64)
  int idx = blockIdx.x * blockDim.x + threadIdx.x;
  if (idx >= N2 * NHEAD * NHOUR) return;
  int l = idx % NHOUR;
  int h = (idx / NHOUR) % NHEAD;
  int n = idx / (NHOUR * NHEAD);
  const float scale = 0.25f;                       // 1/sqrt(16)
  float q[HD];
  const float* qb = qkv + ((size_t)(n * NHOUR + l)) * (3 * DIM) + h * HD;
  for (int d = 0; d < HD; ++d) q[d] = qb[d];
  float lg[NHOUR];
  float mx = -1e30f;
  for (int s = 0; s < NHOUR; ++s) {
    const float* kb = qkv + ((size_t)(n * NHOUR + s)) * (3 * DIM) + DIM + h * HD;
    float dot = 0.f;
    for (int d = 0; d < HD; ++d) dot += q[d] * kb[d];
    dot *= scale;
    lg[s] = dot;
    mx = fmaxf(mx, dot);
  }
  float sum = 0.f;
  for (int s = 0; s < NHOUR; ++s) { lg[s] = expf(lg[s] - mx); sum += lg[s]; }
  float inv = 1.0f / sum;
  float o[HD];
  for (int d = 0; d < HD; ++d) o[d] = 0.f;
  for (int s = 0; s < NHOUR; ++s) {
    float w = lg[s] * inv;
    const float* vb = qkv + ((size_t)(n * NHOUR + s)) * (3 * DIM) + 2 * DIM + h * HD;
    for (int d = 0; d < HD; ++d) o[d] += w * vb[d];
  }
  _Float16* ob = outH + ((size_t)(n * NHOUR + l)) * DIM + h * HD;
  for (int d = 0; d < HD; ++d) ob[d] = (_Float16)o[d];
}

__global__ void ln64_kernel(const float* __restrict__ pre, const float* __restrict__ g,
                            const float* __restrict__ b, float* __restrict__ outF,
                            _Float16* __restrict__ outH, int rows) {
  int r = blockIdx.x * blockDim.x + threadIdx.x;
  if (r >= rows) return;
  const float* x = pre + (size_t)r * DIM;
  float m = 0.f;
  for (int d = 0; d < DIM; ++d) m += x[d];
  m *= (1.0f / DIM);
  float v = 0.f;
  for (int d = 0; d < DIM; ++d) { float t = x[d] - m; v += t * t; }
  v *= (1.0f / DIM);
  float is = rsqrtf(v + 1e-5f);
  for (int d = 0; d < DIM; ++d) {
    float y = (x[d] - m) * is * g[d] + b[d];
    outF[(size_t)r * DIM + d] = y;
    outH[(size_t)r * DIM + d] = (_Float16)y;
  }
}

__global__ void attn_fused_kernel(const float* __restrict__ qp,     // (NREG, 64)
                                  const float* __restrict__ kvp0,   // (NT, 128)
                                  const float* __restrict__ kvp1,   // (NT, 128)
                                  _Float16* __restrict__ outH) {    // (NT, 64)
  int idx = blockIdx.x * blockDim.x + threadIdx.x;
  if (idx >= NT * NHEAD) return;
  int h = idx % NHEAD;
  int tok = idx / NHEAD;
  int n = (tok / NHOUR) % NREG;
  const float* q  = qp   + (size_t)n * DIM + h * HD;
  const float* k0 = kvp0 + (size_t)tok * 128 + h * HD;
  const float* v0 = k0 + DIM;
  const float* k1 = kvp1 + (size_t)tok * 128 + h * HD;
  const float* v1 = k1 + DIM;
  float l0 = 0.f, l1 = 0.f;
  for (int d = 0; d < HD; ++d) { l0 += q[d] * k0[d]; l1 += q[d] * k1[d]; }
  l0 *= 0.25f; l1 *= 0.25f;
  float mx = fmaxf(l0, l1);
  float e0 = expf(l0 - mx), e1 = expf(l1 - mx);
  float inv = 1.0f / (e0 + e1);
  e0 *= inv; e1 *= inv;
  _Float16* ob = outH + (size_t)tok * DIM + h * HD;
  for (int d = 0; d < HD; ++d) ob[d] = (_Float16)(e0 * v0[d] + e1 * v1[d]);
}

__global__ void gate_kernel(const float* __restrict__ mid, const float* __restrict__ condp,
                            _Float16* __restrict__ oH) {
  int idx = blockIdx.x * blockDim.x + threadIdx.x;
  if (idx >= NT * DIM) return;
  int d = idx & 63;
  int tok = idx >> 6;
  int n = (tok / NHOUR) % NREG;
  float g = mid[(size_t)tok * 128 + d]      + condp[(size_t)n * 128 + d];
  float f = mid[(size_t)tok * 128 + 64 + d] + condp[(size_t)n * 128 + 64 + d];
  float s = 1.0f / (1.0f + expf(-g));
  oH[idx] = (_Float16)(s * tanhf(f));
}

__global__ void final_kernel(const float* __restrict__ fin, const float* __restrict__ x_in,
                             float* __restrict__ out) {
  int idx = blockIdx.x * blockDim.x + threadIdx.x;
  if (idx >= NT * DIM) return;
  int d = idx & 63;
  int tok = idx >> 6;
  float res = fin[(size_t)tok * 128 + d];
  float skp = fin[(size_t)tok * 128 + 64 + d];
  out[idx] = (x_in[idx] + res) * 0.70710678118654752f;
  out[(size_t)NT * DIM + idx] = skp;
}

__global__ void cvt_kernel(const float* __restrict__ s, _Float16* __restrict__ d, int n) {
  int i = blockIdx.x * blockDim.x + threadIdx.x;
  if (i < n) d[i] = (_Float16)s[i];
}

// d[b][n*K+k] = s[b][k*N+n]   (transpose + f16)
__global__ void cvtT_kernel(const float* __restrict__ s, _Float16* __restrict__ d,
                            int K, int N, int batch) {
  int idx = blockIdx.x * blockDim.x + threadIdx.x;
  int tot = batch * K * N;
  if (idx >= tot) return;
  int b = idx / (K * N);
  int r = idx % (K * N);
  int k = r / N;
  int n = r % N;
  d[(size_t)b * K * N + (size_t)n * K + k] = (_Float16)s[idx];
}

// ---------------------------------------------------------------------------
// Host orchestration
// ---------------------------------------------------------------------------
extern "C" void kernel_launch(void* const* d_in, const int* in_sizes, int n_in,
                              void* d_out, int out_size, void* d_ws, size_t ws_size,
                              hipStream_t stream) {
  (void)in_sizes; (void)n_in; (void)out_size; (void)ws_size;
  const float* x_in     = (const float*)d_in[0];
  const float* time_emb = (const float*)d_in[1];
  const float* cond     = (const float*)d_in[2];
  const float* KGE      = (const float*)d_in[3];
  const int*   ei       = (const int*)d_in[4];
  const int*   et       = (const int*)d_in[5];
  const float* mlp_w    = (const float*)d_in[6];
  const float* mlp_b    = (const float*)d_in[7];
  const float* kge_w    = (const float*)d_in[8];
  const float* kge_b    = (const float*)d_in[9];
  const float* cond_w   = (const float*)d_in[10];
  const float* cond_b   = (const float*)d_in[11];
  const float* rel_w    = (const float*)d_in[12];
  const float* root_w   = (const float*)d_in[13];
  const float* rgcn_b   = (const float*)d_in[14];
  const float* blk_in_w  = (const float*)d_in[15];
  const float* blk_in_b  = (const float*)d_in[16];
  const float* blk_out_w = (const float*)d_in[17];
  const float* blk_out_b = (const float*)d_in[18];
  const float* ln1_g     = (const float*)d_in[19];
  const float* ln1_b     = (const float*)d_in[20];
  const float* ff1_w     = (const float*)d_in[21];
  const float* ff1_b     = (const float*)d_in[22];
  const float* ff2_w     = (const float*)d_in[23];
  const float* ff2_b     = (const float*)d_in[24];
  const float* ln2_g     = (const float*)d_in[25];
  const float* ln2_b     = (const float*)d_in[26];
  const float* mha_in_w  = (const float*)d_in[27];
  const float* mha_in_b  = (const float*)d_in[28];
  const float* mha_out_w = (const float*)d_in[29];
  const float* mha_out_b = (const float*)d_in[30];
  const float* mid_w     = (const float*)d_in[31];
  const float* mid_b     = (const float*)d_in[32];
  const float* out_w     = (const float*)d_in[33];
  const float* out_b     = (const float*)d_in[34];

  char* w = (char*)d_ws;
  size_t off = 0;
  auto alloc = [&](size_t bytes) -> char* {
    char* p = w + off;
    off += (bytes + 255) & ~(size_t)255;
    return p;
  };

  float*     t_buf  = (float*)alloc((size_t)BSZ * NHOUR * DIM * 4);
  float*     ot     = (float*)alloc((size_t)NT * DIM * 4);
  _Float16*  otH    = (_Float16*)alloc((size_t)NT * DIM * 2);
  _Float16*  xgH    = (_Float16*)alloc((size_t)NT * DIM * 2);
  float*     agg    = (float*)alloc((size_t)GG * NREG * NRel * DIM * 4);
  float*     cnt    = (float*)alloc((size_t)NREG * NRel * 4);
  _Float16*  osH    = (_Float16*)alloc((size_t)NT * DIM * 2);
  float*     bufA   = (float*)alloc((size_t)NT * 256 * 4);   // qkv / kvp0+kvp1 / mid / fin
  float*     bufB   = (float*)alloc((size_t)NT * DIM * 4);   // pre-LN
  _Float16*  halfA  = (_Float16*)alloc((size_t)NT * 256 * 2);
  _Float16*  halfB  = (_Float16*)alloc((size_t)NT * DIM * 2);
  _Float16*  halfC  = (_Float16*)alloc((size_t)NT * DIM * 2);
  float*     qpf    = (float*)alloc((size_t)NREG * DIM * 4);
  float*     condpf = (float*)alloc((size_t)NREG * 128 * 4);
  _Float16*  embH   = (_Float16*)alloc((size_t)NREG * DIM * 2);
  _Float16*  kgeH   = (_Float16*)alloc((size_t)NREG * 32 * 2);
  _Float16*  condH  = (_Float16*)alloc((size_t)NREG * 32 * 2);
  // f16 weight arena
  _Float16* inWH    = (_Float16*)alloc((size_t)2 * 192 * 64 * 2);
  _Float16* outWH   = (_Float16*)alloc((size_t)2 * 64 * 64 * 2);
  _Float16* ff1T    = (_Float16*)alloc((size_t)2 * 256 * 64 * 2);
  _Float16* ff2T    = (_Float16*)alloc((size_t)2 * 64 * 256 * 2);
  _Float16* relT    = (_Float16*)alloc((size_t)NRel * 64 * 64 * 2);
  _Float16* rootT   = (_Float16*)alloc((size_t)64 * 64 * 2);
  _Float16* kgeT    = (_Float16*)alloc((size_t)64 * 32 * 2);
  _Float16* condT   = (_Float16*)alloc((size_t)128 * 32 * 2);
  _Float16* midT    = (_Float16*)alloc((size_t)128 * 64 * 2);
  _Float16* outT    = (_Float16*)alloc((size_t)128 * 64 * 2);
  _Float16* mhaInH  = (_Float16*)alloc((size_t)192 * 64 * 2);
  _Float16* mhaOutH = (_Float16*)alloc((size_t)64 * 64 * 2);

  float* kvp0 = bufA;
  float* kvp1 = bufA + (size_t)NT * 128;

  auto gemm = [&](const _Float16* A, const _Float16* Bt, const float* bias,
                  const float* resid, float* Cf, _Float16* Ch,
                  int M, int N, int K, int relu) {
    dim3 grid(M / (16 * MT), N / 64);
    size_t shmem = (size_t)64 * K * 2;           // Bt panel in LDS
    gemm_f16_kernel<<<grid, dim3(64), shmem, stream>>>(A, Bt, bias, resid, Cf, Ch,
                                                       M, N, K, relu);
  };
  auto cvt = [&](const float* s, _Float16* d, int n) {
    cvt_kernel<<<(n + 255) / 256, 256, 0, stream>>>(s, d, n);
  };
  auto cvtT = [&](const float* s, _Float16* d, int K, int N, int batch) {
    int tot = batch * K * N;
    cvtT_kernel<<<(tot + 255) / 256, 256, 0, stream>>>(s, d, K, N, batch);
  };

  // ---- prep: zero accumulators, f16 weights -------------------------------
  zero_kernel<<<4096, 256, 0, stream>>>(agg, (size_t)GG * NREG * NRel * DIM);
  zero_kernel<<<32, 256, 0, stream>>>(cnt, (size_t)NREG * NRel);
  cvt(blk_in_w, inWH, 2 * 192 * 64);
  cvt(blk_out_w, outWH, 2 * 64 * 64);
  cvt(mha_in_w, mhaInH, 192 * 64);
  cvt(mha_out_w, mhaOutH, 64 * 64);
  cvt(KGE, kgeH, NREG * 32);
  cvt(cond, condH, NREG * 32);
  cvtT(rel_w, relT, 64, 64, NRel);
  cvtT(root_w, rootT, 64, 64, 1);
  cvtT(ff1_w, ff1T, 64, 256, 2);
  cvtT(ff2_w, ff2T, 256, 64, 2);
  cvtT(kge_w, kgeT, 32, 64, 1);
  cvtT(cond_w, condT, 32, 128, 1);
  cvtT(mid_w, midT, 64, 128, 1);
  cvtT(out_w, outT, 64, 128, 1);

  // ---- time embedding + broadcast add -------------------------------------
  tmlp_kernel<<<(BSZ * NHOUR * DIM + 127) / 128, 128, 0, stream>>>(time_emb, mlp_w, mlp_b, t_buf);
  add_time_kernel<<<(NT * DIM) / 256, 256, 0, stream>>>(x_in, t_buf, ot, otH, xgH);

  // ---- RGCN: count, scatter-add, normalize, fused WMMA GEMM ---------------
  cnt_kernel<<<NE / 256, 256, 0, stream>>>(ei, et, cnt);
  scatter_kernel<<<NE, 256, 0, stream>>>(ei, et, ot, agg);
  {
    size_t tot = (size_t)GG * NREG * NRel * DIM;
    norm_kernel<<<(unsigned)((tot + 255) / 256), 256, 0, stream>>>(agg, cnt);
  }
  rgcn_gemm_kernel<<<GG * NREG / 16, 128, 0, stream>>>(agg, xgH, relT, rootT, rgcn_b, osH);

  // ---- temporal transformer blocks ----------------------------------------
  for (int i = 0; i < 2; ++i) {
    const _Float16* inW_i  = inWH  + (size_t)i * 192 * 64;
    const _Float16* outW_i = outWH + (size_t)i * 64 * 64;
    const _Float16* f1T_i  = ff1T  + (size_t)i * 256 * 64;
    const _Float16* f2T_i  = ff2T  + (size_t)i * 64 * 256;
    // qkv projection (N=192)
    gemm(otH, inW_i, blk_in_b + i * 192, nullptr, bufA, nullptr, NT, 192, 64, 0);
    attn_temporal_kernel<<<(N2 * NHEAD * NHOUR) / 256, 256, 0, stream>>>(bufA, halfB);
    // out projection + residual -> pre-LN
    gemm(halfB, outW_i, blk_out_b + i * 64, ot, bufB, nullptr, NT, 64, 64, 0);
    ln64_kernel<<<(NT + 255) / 256, 256, 0, stream>>>(bufB, ln1_g + i * 64, ln1_b + i * 64, ot, otH, NT);
    // FFN
    gemm(otH, f1T_i, ff1_b + i * 256, nullptr, nullptr, halfA, NT, 256, 64, 1);
    gemm(halfA, f2T_i, ff2_b + i * 64, ot, bufB, nullptr, NT, 64, 256, 0);
    ln64_kernel<<<(NT + 255) / 256, 256, 0, stream>>>(bufB, ln2_g + i * 64, ln2_b + i * 64, ot, otH, NT);
  }

  // ---- fused cross-attention ----------------------------------------------
  gemm(kgeH, kgeT, kge_b, nullptr, nullptr, embH, NREG, 64, 32, 0);                 // emb
  gemm(embH, mhaInH, mha_in_b, nullptr, qpf, nullptr, NREG, 64, 64, 0);             // q proj
  gemm(osH, mhaInH + 64 * 64, mha_in_b + 64, nullptr, kvp0, nullptr, NT, 128, 64, 0); // k|v spatial
  gemm(otH, mhaInH + 64 * 64, mha_in_b + 64, nullptr, kvp1, nullptr, NT, 128, 64, 0); // k|v temporal
  attn_fused_kernel<<<(NT * NHEAD) / 256, 256, 0, stream>>>(qpf, kvp0, kvp1, halfB);
  gemm(halfB, mhaOutH, mha_out_b, nullptr, nullptr, halfC, NT, 64, 64, 0);          // out proj

  // ---- gating + output head -----------------------------------------------
  gemm(condH, condT, cond_b, nullptr, condpf, nullptr, NREG, 128, 32, 0);
  gemm(halfC, midT, mid_b, nullptr, bufA, nullptr, NT, 128, 64, 0);                 // mid
  gate_kernel<<<(NT * DIM) / 256, 256, 0, stream>>>(bufA, condpf, halfB);
  gemm(halfB, outT, out_b, nullptr, bufA, nullptr, NT, 128, 64, 0);                 // fin
  final_kernel<<<(NT * DIM) / 256, 256, 0, stream>>>(bufA, x_in, (float*)d_out);
}
